// RNN_Model_51548197487201
// MI455X (gfx1250) — compile-verified, matrix operand order checked by reading, not run on hardware
//
#include <hip/hip_runtime.h>
#include <hip/hip_bf16.h>
#include <math.h>

// Problem constants (from reference)
#define B_     64
#define T_     2048
#define VOCAB_ 32
#define D_     128
#define H_     256
#define NCLS_  32

// LDS h-state row stride in bf16 elements: 264*2 = 528 bytes = 33*16 (B128-aligned),
// 528/4 = 132 -> 4-bank rotation per row -> conflict-free ds_load_b128 across 16 rows.
#define LDS_STRIDE 264

typedef __attribute__((ext_vector_type(16))) __bf16 v16bf;
typedef __attribute__((ext_vector_type(8)))  __bf16 v8bf;
typedef __attribute__((ext_vector_type(8)))  float  v8f;

static __device__ __forceinline__ v16bf frag_cat(v8bf lo, v8bf hi) {
  return __builtin_shufflevector(lo, hi, 0,1,2,3,4,5,6,7,8,9,10,11,12,13,14,15);
}

// 16-element bf16 WMMA operand fragment from a row-major bf16 row:
// elems 0..7 = row[k0..k0+7], elems 8..15 = row[k0+16..k0+23]
// (ISA 16-bit A/B layout; caller supplies k0 = 32*chunk + (laneHi ? 8 : 0)).
static __device__ __forceinline__ v16bf frag_row_bf16(const __bf16* row, int k0) {
  v8bf lo = *(const v8bf*)(row + k0);
  v8bf hi = *(const v8bf*)(row + k0 + 16);
  return frag_cat(lo, hi);
}

// Same fragment, converting from f32 source (used once, to cache W_hh in VGPRs).
static __device__ __forceinline__ v16bf frag_row_f32(const float* row, int k0) {
  v16bf f;
#pragma unroll
  for (int i = 0; i < 8; ++i) {
    f[i]     = (__bf16)row[k0 + i];
    f[i + 8] = (__bf16)row[k0 + 16 + i];
  }
  return f;
}

static __device__ __forceinline__ v8f wmma_bf16(v16bf a, v16bf b, v8f c) {
  return __builtin_amdgcn_wmma_f32_16x16x32_bf16(false, a, false, b, (short)0, c,
                                                 false, false);
}

// Fast tanh on the recurrence critical path: prefer the gfx1250 TRANS op,
// else 4-op exp2/rcp form (saturates correctly: exp2->inf => 1, exp2->0 => -1).
static __device__ __forceinline__ float fast_tanh(float x) {
#if __has_builtin(__builtin_amdgcn_tanhf)
  return __builtin_amdgcn_tanhf(x);
#else
  float e = __builtin_amdgcn_exp2f(x * 2.885390081777927f); // 2*log2(e)
  return 1.0f - 2.0f * __builtin_amdgcn_rcpf(e + 1.0f);
#endif
}

// ---------------------------------------------------------------------------
// embW[v][n] = (b_ih0+b_hh0)[n] + sum_d emb[v][d] * W_ih0[n][d]   (32x256, tiny)
// ---------------------------------------------------------------------------
__global__ void embw_kernel(const float* __restrict__ emb,
                            const float* __restrict__ W_ih0,
                            const float* __restrict__ b_ih0,
                            const float* __restrict__ b_hh0,
                            float* __restrict__ embw) {
  int i = blockIdx.x * blockDim.x + threadIdx.x;
  if (i >= VOCAB_ * H_) return;
  int v = i / H_, n = i % H_;
  float s = b_ih0[n] + b_hh0[n];
#pragma unroll 4
  for (int d = 0; d < D_; ++d) s += emb[v * D_ + d] * W_ih0[n * D_ + d];
  embw[i] = s;
}

// f32 -> bf16 bulk convert (W_ih1 -> bf16 once so the big GEMM reads bf16 from L2)
__global__ void cvt_bf16_kernel(const float* __restrict__ src,
                                __bf16* __restrict__ dst, int n) {
  int i = blockIdx.x * blockDim.x + threadIdx.x;
  if (i < n) dst[i] = (__bf16)src[i];
}

// ---------------------------------------------------------------------------
// Sequential tanh-RNN scan. One workgroup per 16-row batch slab (grid = 4),
// 16 waves; wave w owns output columns [16w, 16w+16). W_hh lives in VGPRs as
// 8 pre-packed bf16 B-fragments for the whole scan; h-state ping-pongs between
// two LDS buffers => ONE barrier per step. Per step per wave:
// 8x v_wmma_f32_16x16x32_bf16 on two accumulators (chain len 4), hw tanh,
// next-step xp loads software-pipelined one iteration ahead.
// ---------------------------------------------------------------------------
template <bool IS_L0>
__global__ __launch_bounds__(512, 1)
void rnn_scan_kernel(const int* __restrict__ X,        // [B][T]   (L0)
                     const float* __restrict__ embw,   // [32][H]  (L0)
                     const __bf16* __restrict__ xp_bf, // [B][T][H](L1)
                     const float* __restrict__ W_hh,   // [H][H] f32
                     __bf16* __restrict__ hs_out,      // [B][T][H](L0)
                     float* __restrict__ pooled)       // [B][H]   (L1, time-sum)
{
  __shared__ __align__(16) __bf16 hbuf[2][16 * LDS_STRIDE];
  __shared__ float embw_s[VOCAB_ * H_];

  const int tid   = threadIdx.x;
  const int lane  = tid & 31;
  const int wave  = tid >> 5;        // 0..15
  const int lmod  = lane & 15;
  const int lhalf = lane >> 4;       // 0: lanes 0-15, 1: lanes 16-31
  const int n     = wave * 16 + lmod;   // this lane's output column
  const int b0    = blockIdx.x * 16;    // batch slab base
  const int koff  = lhalf * 8;

  if (IS_L0) {
    for (int i = tid; i < VOCAB_ * H_; i += 512) embw_s[i] = embw[i];
  }
  for (int i = tid; i < 16 * LDS_STRIDE; i += 512) {
    hbuf[0][i] = (__bf16)0.0f;
    hbuf[1][i] = (__bf16)0.0f;
  }

  // Cache W_hh^T as 8 B-fragments: B[k][n] = W_hh[n][k] -> contiguous row reads.
  v16bf Bfrag[8];
  {
    const float* wrow = W_hh + (size_t)n * H_;
#pragma unroll
    for (int c = 0; c < 8; ++c) Bfrag[c] = frag_row_f32(wrow, 32 * c + koff);
  }

  // xp loader (L0: token -> folded embW table in LDS; L1: precomputed bf16 XP1)
  auto load_xp = [&](int t) {
    v8f v;
#pragma unroll
    for (int r = 0; r < 8; ++r) {
      const int b = b0 + r + lhalf * 8;
      if (IS_L0) {
        int tok = X[(size_t)b * T_ + t];
        v[r] = embw_s[tok * H_ + n];
      } else {
        v[r] = (float)xp_bf[((size_t)b * T_ + t) * H_ + n];
      }
    }
    return v;
  };

  v8f pool = {};
  __syncthreads();

  v8f xp_cur = load_xp(0);

  for (int t = 0; t < T_; ++t) {
    // issue next step's xp loads early so they overlap WMMA + tanh
    const int tn = (t + 1 < T_) ? t + 1 : t;
    v8f xp_next = load_xp(tn);

    // acc += h_{t-1} @ W_hh^T  (A from LDS ping buffer, B resident in VGPRs)
    const __bf16* arow = hbuf[t & 1] + lmod * LDS_STRIDE;
    v8f acc0 = xp_cur;
    v8f acc1 = {};
#pragma unroll
    for (int c = 0; c < 8; c += 2) {
      v16bf a0 = frag_row_bf16(arow, 32 * c + koff);
      v16bf a1 = frag_row_bf16(arow, 32 * (c + 1) + koff);
      acc0 = wmma_bf16(a0, Bfrag[c],     acc0);
      acc1 = wmma_bf16(a1, Bfrag[c + 1], acc1);
    }

    float h[8];
#pragma unroll
    for (int i = 0; i < 8; ++i) h[i] = fast_tanh(acc0[i] + acc1[i]);

    // write h_t into the pong buffer (no conflict with this step's reads)
    __bf16* nxt = hbuf[(t + 1) & 1];
#pragma unroll
    for (int r = 0; r < 8; ++r) {
      const int m = r + lhalf * 8;
      __bf16 hv = (__bf16)h[r];
      nxt[m * LDS_STRIDE + n] = hv;
      if (IS_L0) hs_out[((size_t)(b0 + m) * T_ + t) * H_ + n] = hv;
    }
    if (!IS_L0) {
#pragma unroll
      for (int r = 0; r < 8; ++r) pool[r] += h[r];
    }
    xp_cur = xp_next;
    __syncthreads();   // single barrier: h_t visible, old reads retired
  }

  if (!IS_L0) {
#pragma unroll
    for (int r = 0; r < 8; ++r) {
      const int m = r + lhalf * 8;
      pooled[(size_t)(b0 + m) * H_ + n] = pool[r];
    }
  }
}

// ---------------------------------------------------------------------------
// XP1 = HS0 @ W_ih1^T + (b_ih1+b_hh1): M=131072, N=256, K=256, bf16 WMMA.
// Each wave keeps its 8 W_ih1 B-fragments in VGPRs and sweeps 4 M-tiles.
// ---------------------------------------------------------------------------
#define MT_PER_WAVE 4
__global__ __launch_bounds__(256)
void xp1_gemm_kernel(const __bf16* __restrict__ hs0,   // [B*T][H]
                     const __bf16* __restrict__ w1bf,  // [H][H] = W_ih1 bf16
                     const float* __restrict__ b_ih1,
                     const float* __restrict__ b_hh1,
                     __bf16* __restrict__ xp1)         // [B*T][H]
{
  const int lane  = threadIdx.x & 31;
  const int wave  = threadIdx.x >> 5;
  const int lmod  = lane & 15;
  const int lhalf = lane >> 4;
  const int koff  = lhalf * 8;
  const int gw    = blockIdx.x * 8 + wave;   // 0..32767
  const int ntile = gw & 15;
  const int mt0   = (gw >> 4) * MT_PER_WAVE; // first of 4 M-tiles

  const int n = ntile * 16 + lmod;           // B col / C col
  const float bias = b_ih1[n] + b_hh1[n];

  v16bf Bf[8];
  {
    const __bf16* brow = w1bf + (size_t)n * H_;
#pragma unroll
    for (int c = 0; c < 8; ++c) Bf[c] = frag_row_bf16(brow, 32 * c + koff);
  }

  for (int mi = 0; mi < MT_PER_WAVE; ++mi) {
    const int mtile = mt0 + mi;
    const __bf16* arow = hs0 + (size_t)(mtile * 16 + lmod) * H_;
    v8f acc = {};
#pragma unroll
    for (int c = 0; c < 8; ++c) {
      v16bf a = frag_row_bf16(arow, 32 * c + koff);
      acc = wmma_bf16(a, Bf[c], acc);
    }
#pragma unroll
    for (int r = 0; r < 8; ++r) {
      const int mr = mtile * 16 + r + lhalf * 8;
      xp1[(size_t)mr * H_ + n] = (__bf16)(acc[r] + bias);
    }
  }
}

// ---------------------------------------------------------------------------
// out[b][c] = (pooled_sum[b] / T) . W_out[c] + b_out[c]   (64x32, tiny)
// ---------------------------------------------------------------------------
__global__ void head_kernel(const float* __restrict__ pooled,
                            const float* __restrict__ W_out,
                            const float* __restrict__ b_out,
                            float* __restrict__ out) {
  int i = blockIdx.x * blockDim.x + threadIdx.x;
  if (i >= B_ * NCLS_) return;
  int b = i / NCLS_, c = i % NCLS_;
  float s = 0.f;
#pragma unroll 4
  for (int k = 0; k < H_; ++k) s += pooled[b * H_ + k] * W_out[c * H_ + k];
  out[i] = s * (1.0f / (float)T_) + b_out[c];
}

// ---------------------------------------------------------------------------
extern "C" void kernel_launch(void* const* d_in, const int* in_sizes, int n_in,
                              void* d_out, int out_size, void* d_ws, size_t ws_size,
                              hipStream_t stream) {
  (void)in_sizes; (void)n_in; (void)out_size; (void)ws_size;
  const int*   X      = (const int*)  d_in[0];
  const float* emb    = (const float*)d_in[1];
  const float* W_ih0  = (const float*)d_in[2];
  const float* W_hh0  = (const float*)d_in[3];
  const float* b_ih0  = (const float*)d_in[4];
  const float* b_hh0  = (const float*)d_in[5];
  const float* W_ih1  = (const float*)d_in[6];
  const float* W_hh1  = (const float*)d_in[7];
  const float* b_ih1  = (const float*)d_in[8];
  const float* b_hh1  = (const float*)d_in[9];
  const float* W_out  = (const float*)d_in[10];
  const float* b_out  = (const float*)d_in[11];
  float* out = (float*)d_out;

  // Workspace layout (~134.5 MB)
  char* p = (char*)d_ws;
  __bf16* HS0    = (__bf16*)p; p += (size_t)B_ * T_ * H_ * 2;   // 64 MB
  __bf16* XP1    = (__bf16*)p; p += (size_t)B_ * T_ * H_ * 2;   // 64 MB
  float*  embw   = (float*) p; p += (size_t)VOCAB_ * H_ * 4;    // 32 KB
  __bf16* W1bf   = (__bf16*)p; p += (size_t)H_ * H_ * 2;        // 128 KB
  float*  pooled = (float*) p; p += (size_t)B_ * H_ * 4;        // 64 KB

  // 1) fold embedding through L0 input projection (+biases) into a 32x256 table
  embw_kernel<<<(VOCAB_ * H_ + 255) / 256, 256, 0, stream>>>(emb, W_ih0, b_ih0,
                                                             b_hh0, embw);
  // 2) W_ih1 -> bf16 once
  cvt_bf16_kernel<<<(H_ * H_ + 255) / 256, 256, 0, stream>>>(W_ih1, W1bf, H_ * H_);
  // 3) layer-0 scan (4 slabs x 16 waves), emits HS0 bf16
  rnn_scan_kernel<true><<<B_ / 16, 512, 0, stream>>>(X, embw, nullptr, W_hh0,
                                                     HS0, nullptr);
  // 4) bulk bf16 WMMA GEMM: XP1 = HS0 @ W_ih1^T + bias (4 M-tiles per wave)
  {
    const int total_waves = (B_ * T_ / 16 / MT_PER_WAVE) * 16; // 32768
    xp1_gemm_kernel<<<total_waves / 8, 256, 0, stream>>>(HS0, W1bf, b_ih1,
                                                         b_hh1, XP1);
  }
  // 5) layer-1 scan, accumulating the time-sum for mean pooling
  rnn_scan_kernel<false><<<B_ / 16, 512, 0, stream>>>(nullptr, nullptr, XP1,
                                                      W_hh1, nullptr, pooled);
  // 6) classifier head
  head_kernel<<<(B_ * NCLS_ + 255) / 256, 256, 0, stream>>>(pooled, W_out, b_out,
                                                            out);
}
